// RNNDecoderRL_59433757442508
// MI455X (gfx1250) — matrix operand order repaired and checked by reading, not compiled, not run on hardware
//
#include <hip/hip_runtime.h>

// ---------------------------------------------------------------------------
// Types for CDNA5 WMMA (wave32, v_wmma_f32_16x16x32_bf16)
// ---------------------------------------------------------------------------
typedef __bf16 bf16;
typedef __attribute__((ext_vector_type(16))) __bf16    v16bf;
typedef __attribute__((ext_vector_type(8)))  float     v8f;
typedef __attribute__((ext_vector_type(4)))  unsigned  v4u;

union Frag16 { v16bf bf; v4u u[2]; };

// One k-step worth of operands: 4 A-tiles (64 rows) x 2 B-tiles (32 cols)
struct FragSet { Frag16 a[4]; Frag16 b0, b1; };

// Model constants
#define Bsz 128
#define Tt  20
#define Vv  20000
#define DIc 512
#define Hh  1024
#define G4  4096
#define NI  36
#define NP  100

__device__ __forceinline__ void load_set(FragSet& s, const bf16* ap0, size_t astep,
                                         const bf16* bp0, const bf16* bp1, int k0) {
    s.b0.u[0] = *(const v4u*)(bp0 + k0);
    s.b0.u[1] = *(const v4u*)(bp0 + k0 + 8);
    s.b1.u[0] = *(const v4u*)(bp1 + k0);
    s.b1.u[1] = *(const v4u*)(bp1 + k0 + 8);
#pragma unroll
    for (int i = 0; i < 4; ++i) {
        const bf16* ap = ap0 + (size_t)i * astep + k0;
        s.a[i].u[0] = *(const v4u*)(ap);        // K = k0 + hh*16 .. +7
        s.a[i].u[1] = *(const v4u*)(ap + 8);    // K = k0 + hh*16 + 8 .. +15
    }
}

__device__ __forceinline__ void consume_set(const FragSet& s, v8f acc[4][2]) {
#pragma unroll
    for (int i = 0; i < 4; ++i) {
        acc[i][0] = __builtin_amdgcn_wmma_f32_16x16x32_bf16(
            false, s.a[i].bf, false, s.b0.bf, (short)0, acc[i][0], false, false);
        acc[i][1] = __builtin_amdgcn_wmma_f32_16x16x32_bf16(
            false, s.a[i].bf, false, s.b1.bf, (short)0, acc[i][1], false, false);
    }
}

// ---------------------------------------------------------------------------
// GEMM: C[M,N] = A[M,K](bf16,row-major,lda) x BT[N,K](bf16,row-major) + bias[N]
// Each wave computes a 64x32 output block: 4 M-tiles x 2 N-tiles.
// Per k-step: 12 batched b128 loads, then 8 independent WMMAs.
// Requires: M % 64 == 0, N % 32 == 0, K % 32 == 0 (true for all calls here).
// ---------------------------------------------------------------------------
__global__ void gemm_bf16_kernel(const bf16* __restrict__ A, int lda,
                                 const bf16* __restrict__ BT,
                                 const float* __restrict__ bias,
                                 float* __restrict__ C, int ldc,
                                 int N, int K)
{
    const int wave   = threadIdx.x >> 5;
    const int lane   = threadIdx.x & 31;
    const int tileNg = blockIdx.x * (blockDim.x >> 5) + wave;   // 32-col group
    if (tileNg * 32 >= N) return;

    const int l  = lane & 15;      // row/col within 16x16 tile
    const int hh = lane >> 4;      // lane-half selects K sub-range
    const int n0 = tileNg * 32 + l;
    const int n1 = n0 + 16;
    const int mbase = blockIdx.y * 64;

    v8f acc[4][2] = {};

    const bf16* bp0 = BT + (size_t)n0 * K + hh * 16;
    const bf16* bp1 = BT + (size_t)n1 * K + hh * 16;
    const bf16* ap0 = A + (size_t)(mbase + l) * lda + hh * 16;
    const size_t astep = (size_t)16 * lda;

    for (int k0 = 0; k0 < K; k0 += 32) {
        FragSet s;
        load_set(s, ap0, astep, bp0, bp1, k0);
        consume_set(s, acc);
    }

    const float bv0 = bias ? bias[n0] : 0.0f;
    const float bv1 = bias ? bias[n1] : 0.0f;
#pragma unroll
    for (int i = 0; i < 4; ++i) {
        const int r0 = mbase + i * 16 + hh * 8;   // C/D layout: VGPR j -> M = half*8 + j
#pragma unroll
        for (int j = 0; j < 8; ++j) {
            C[(size_t)(r0 + j) * ldc + n0] = acc[i][0][j] + bv0;
            C[(size_t)(r0 + j) * ldc + n1] = acc[i][1][j] + bv1;
        }
    }
}

// ---------------------------------------------------------------------------
// Small helper kernels
// ---------------------------------------------------------------------------
__global__ void zero_u32_kernel(unsigned* p, int n) {
    int i = blockIdx.x * blockDim.x + threadIdx.x;
    if (i < n) p[i] = 0u;
}

__global__ void f32_to_bf16_kernel(const float* __restrict__ s, bf16* __restrict__ d, int n) {
    int i = blockIdx.x * blockDim.x + threadIdx.x;
    if (i < n) d[i] = (bf16)s[i];
}

// dst[N,K] bf16 = transpose of src[K,N] f32
__global__ void transpose_to_bf16_kernel(const float* __restrict__ s, bf16* __restrict__ d,
                                         int K, int N) {
    int i = blockIdx.x * blockDim.x + threadIdx.x;
    if (i < N * K) {
        int nn = i / K, kk = i - nn * K;
        d[i] = (bf16)s[(size_t)kk * N + nn];
    }
}

// dst[N, k0+k1] bf16 = [ s0[N,k0] | s1[N,k1] ]  (fused Wih|Whh)
__global__ void pack2_bf16_kernel(const float* __restrict__ s0, int k0,
                                  const float* __restrict__ s1, int k1,
                                  bf16* __restrict__ d, int N) {
    int kw = k0 + k1;
    int i = blockIdx.x * blockDim.x + threadIdx.x;
    if (i < N * kw) {
        int nn = i / kw, kk = i - nn * kw;
        float v = (kk < k0) ? s0[(size_t)nn * k0 + kk] : s1[(size_t)nn * k1 + (kk - k0)];
        d[i] = (bf16)v;
    }
}

__global__ void concat2_f32_kernel(const float* a, int na, const float* b, int nb, float* d) {
    int i = blockIdx.x * blockDim.x + threadIdx.x;
    if (i < na + nb) d[i] = (i < na) ? a[i] : b[i - na];
}

// Embedding gather + ReLU -> x_vis[:, 512:1024] bf16
__global__ void embed_kernel(const int* __restrict__ q, const float* __restrict__ emb,
                             bf16* __restrict__ xvis, int t) {
    int b = blockIdx.x;
    int tok = (t == 0) ? 1 : q[b * Tt + (t - 1)];   // SOS = 1
    for (int d = threadIdx.x; d < 512; d += blockDim.x) {
        float e = emb[(size_t)tok * 512 + d];
        xvis[(size_t)b * 2048 + 512 + d] = (bf16)(e > 0.f ? e : 0.f);
    }
}

// LSTM elementwise: gates g[B,4096] (i,f,g,o), cell c[B,1024]; h -> bf16 slices
__global__ void lstm_elem_kernel(const float* __restrict__ g, float* __restrict__ c,
                                 bf16* d0, int ld0, bf16* d1, int ld1) {
    int i = blockIdx.x * blockDim.x + threadIdx.x;
    if (i >= Bsz * Hh) return;
    int b = i >> 10, j = i & 1023;
    const float* gb = g + (size_t)b * G4;
    float ig = gb[j], fg = gb[Hh + j], gg = gb[2 * Hh + j], og = gb[3 * Hh + j];
    float si = 1.f / (1.f + expf(-ig));
    float sf = 1.f / (1.f + expf(-fg));
    float so = 1.f / (1.f + expf(-og));
    float cc = sf * c[i] + si * tanhf(gg);
    float hv = so * tanhf(cc);
    c[i] = cc;
    d0[(size_t)b * ld0 + j] = (bf16)hv;
    if (d1) d1[(size_t)b * ld1 + j] = (bf16)hv;
}

// Additive-attention scores (img N=36 and ppl N=100 fused), H=1024.
// qW[B,2048] already has the attention bias folded in via GEMM bias.
__global__ void scores_kernel(const float* __restrict__ qW,
                              const float* __restrict__ mI, const float* __restrict__ mP,
                              const float* __restrict__ vI, const float* __restrict__ vP,
                              float* __restrict__ sc) {
    int p = blockIdx.x;
    int b = p / (NI + NP), n = p - b * (NI + NP);
    const float *q, *m, *v;
    if (n < NI) { q = qW + (size_t)b * 2048;        m = mI + ((size_t)b * NI + n) * Hh;        v = vI; }
    else        { q = qW + (size_t)b * 2048 + 1024; m = mP + ((size_t)b * NP + (n - NI)) * Hh; v = vP; }
    __shared__ float red[256];
    float s = 0.f;
    for (int h = threadIdx.x; h < Hh; h += 256) s += tanhf(q[h] + m[h]) * v[h];
    red[threadIdx.x] = s; __syncthreads();
    for (int k = 128; k > 0; k >>= 1) {
        if (threadIdx.x < k) red[threadIdx.x] += red[threadIdx.x + k];
        __syncthreads();
    }
    if (threadIdx.x == 0) sc[p] = red[0];
}

// Pool-attention scores, H=512
__global__ void pool_scores_kernel(const float* __restrict__ qWp, const float* __restrict__ m,
                                   const float* __restrict__ v, float* __restrict__ sc) {
    int p = blockIdx.x;
    int b = p / NI, n = p - b * NI;
    const float* q = qWp + (size_t)b * 512;
    const float* mm = m + ((size_t)b * NI + n) * 512;
    __shared__ float red[256];
    float s = 0.f;
    for (int h = threadIdx.x; h < 512; h += 256) s += tanhf(q[h] + mm[h]) * v[h];
    red[threadIdx.x] = s; __syncthreads();
    for (int k = 128; k > 0; k >>= 1) {
        if (threadIdx.x < k) red[threadIdx.x] += red[threadIdx.x + k];
        __syncthreads();
    }
    if (threadIdx.x == 0) sc[p] = red[0];
}

// In-place row softmax: row b at base + b*stride, length n (n <= 128 here)
__global__ void softmax_kernel(float* __restrict__ base, int stride, int n) {
    int b = blockIdx.x, tid = threadIdx.x;
    float* s = base + (size_t)b * stride;
    __shared__ float red[128];
    float mx = -1e30f;
    for (int i = tid; i < n; i += 128) mx = fmaxf(mx, s[i]);
    red[tid] = mx; __syncthreads();
    for (int k = 64; k > 0; k >>= 1) { if (tid < k) red[tid] = fmaxf(red[tid], red[tid + k]); __syncthreads(); }
    mx = red[0]; __syncthreads();
    float sum = 0.f;
    for (int i = tid; i < n; i += 128) sum += expf(s[i] - mx);
    red[tid] = sum; __syncthreads();
    for (int k = 64; k > 0; k >>= 1) { if (tid < k) red[tid] += red[tid + k]; __syncthreads(); }
    float inv = 1.f / red[0]; __syncthreads();
    for (int i = tid; i < n; i += 128) s[i] = expf(s[i] - mx) * inv;
}

// Pooled context -> x_vis[:, 0:512] bf16
__global__ void ctx_pool_kernel(const float* __restrict__ sc, const float* __restrict__ feats,
                                bf16* __restrict__ xvis) {
    int b = blockIdx.x;
    for (int d = threadIdx.x; d < 512; d += blockDim.x) {
        float s = 0.f;
        for (int n = 0; n < NI; ++n)
            s += sc[b * NI + n] * feats[((size_t)b * NI + n) * DIc + d];
        xvis[(size_t)b * 2048 + d] = (bf16)s;
    }
}

// img_ctx + ppl_ctx -> x_lang[:, 0:512] bf16
__global__ void ctx_main_kernel(const float* __restrict__ sc,
                                const float* __restrict__ imgf, const float* __restrict__ pplf,
                                bf16* __restrict__ xlang) {
    int b = blockIdx.x;
    for (int d = threadIdx.x; d < 512; d += blockDim.x) {
        float s = 0.f;
        for (int n = 0; n < NI; ++n)
            s += sc[b * (NI + NP) + n] * imgf[((size_t)b * NI + n) * DIc + d];
        for (int n = 0; n < NP; ++n)
            s += sc[b * (NI + NP) + NI + n] * pplf[((size_t)b * NP + n) * DIc + d];
        xlang[(size_t)b * 2560 + d] = (bf16)s;
    }
}

// log_softmax over V=20000 + argmax; writes into d_out
__global__ void logsoftmax_kernel(const float* __restrict__ logits, float* __restrict__ out, int t) {
    int b = blockIdx.x, tid = threadIdx.x;
    const float* L = logits + (size_t)b * Vv;
    __shared__ float rmax[256];
    __shared__ int   rarg[256];
    float mx = -1e30f; int am = 0;
    for (int v = tid; v < Vv; v += 256) { float x = L[v]; if (x > mx) { mx = x; am = v; } }
    rmax[tid] = mx; rarg[tid] = am; __syncthreads();
    for (int k = 128; k > 0; k >>= 1) {
        if (tid < k && rmax[tid + k] > rmax[tid]) { rmax[tid] = rmax[tid + k]; rarg[tid] = rarg[tid + k]; }
        __syncthreads();
    }
    mx = rmax[0]; am = rarg[0]; __syncthreads();
    float sum = 0.f;
    for (int v = tid; v < Vv; v += 256) sum += expf(L[v] - mx);
    rmax[tid] = sum; __syncthreads();
    for (int k = 128; k > 0; k >>= 1) { if (tid < k) rmax[tid] += rmax[tid + k]; __syncthreads(); }
    float lse = mx + logf(rmax[0]);
    size_t base = ((size_t)b * Tt + t) * (size_t)Vv;
    for (int v = tid; v < Vv; v += 256) out[base + v] = L[v] - lse;
    if (tid == 0) out[(size_t)Bsz * Tt * Vv + b * Tt + t] = (float)am;
}

// ---------------------------------------------------------------------------
// Host side
// ---------------------------------------------------------------------------
static inline void launch_gemm(hipStream_t s, const bf16* A, int lda, const bf16* BT,
                               const float* bias, float* C, int ldc, int M, int N, int K) {
    dim3 grid((N / 32 + 7) / 8, M / 64);   // wave = 64x32 output block
    hipLaunchKernelGGL(gemm_bf16_kernel, grid, dim3(256), 0, s, A, lda, BT, bias, C, ldc, N, K);
}
static inline dim3 g1(int n) { return dim3((n + 255) / 256); }

extern "C" void kernel_launch(void* const* d_in, const int* in_sizes, int n_in,
                              void* d_out, int out_size, void* d_ws, size_t ws_size,
                              hipStream_t stream) {
    const float* img_feats = (const float*)d_in[0];
    const float* ppl_feats = (const float*)d_in[1];
    const float* answer    = (const float*)d_in[2];
    const int*   question  = (const int*)  d_in[3];
    const float* emb       = (const float*)d_in[4];
    const float* pool_Wq   = (const float*)d_in[5];
    const float* pool_Wm   = (const float*)d_in[6];
    const float* pool_b    = (const float*)d_in[7];
    const float* pool_v    = (const float*)d_in[8];
    const float* img_Wq    = (const float*)d_in[9];
    const float* img_Wm    = (const float*)d_in[10];
    const float* img_b     = (const float*)d_in[11];
    const float* img_v     = (const float*)d_in[12];
    const float* ppl_Wq    = (const float*)d_in[13];
    const float* ppl_Wm    = (const float*)d_in[14];
    const float* ppl_b     = (const float*)d_in[15];
    const float* ppl_v     = (const float*)d_in[16];
    const float* vis_Wih   = (const float*)d_in[17];
    const float* vis_Whh   = (const float*)d_in[18];
    const float* vis_b     = (const float*)d_in[19];
    const float* lang_Wih  = (const float*)d_in[20];
    const float* lang_Whh  = (const float*)d_in[21];
    const float* lang_b    = (const float*)d_in[22];
    const float* proj_W    = (const float*)d_in[23];
    const float* proj_b    = (const float*)d_in[24];
    float* out = (float*)d_out;

    // ---- workspace carve-out (256B aligned) ----
    char* p = (char*)(((uintptr_t)d_ws + 255) & ~(uintptr_t)255);
    auto alloc = [&](size_t bytes) { void* r = p; p += (bytes + 255) & ~(size_t)255; return r; };

    bf16* wc_vis   = (bf16*)alloc((size_t)G4 * 2048 * 2);   // [4096,2048] = Wih|Whh
    bf16* wc_lang  = (bf16*)alloc((size_t)G4 * 2560 * 2);   // [4096,2560]
    bf16* w_projT  = (bf16*)alloc((size_t)Vv * Hh * 2);     // [20000,1024]
    bf16* w_qT     = (bf16*)alloc((size_t)2048 * Hh * 2);   // [2048,1024] = imgWq^T|pplWq^T
    bf16* w_plWqT  = (bf16*)alloc((size_t)512 * 512 * 2);
    bf16* w_plWmT  = (bf16*)alloc((size_t)512 * 512 * 2);
    bf16* w_imWmT  = (bf16*)alloc((size_t)Hh * 512 * 2);
    bf16* w_ppWmT  = (bf16*)alloc((size_t)Hh * 512 * 2);
    bf16* a_img    = (bf16*)alloc((size_t)Bsz * NI * DIc * 2);
    bf16* a_ppl    = (bf16*)alloc((size_t)Bsz * NP * DIc * 2);
    bf16* a_hint   = (bf16*)alloc((size_t)Bsz * 512 * 2);
    float* qb      = (float*)alloc(2048 * 4);               // img_b|ppl_b
    float* mWm_pl  = (float*)alloc((size_t)Bsz * NI * 512 * 4);
    float* mWm_im  = (float*)alloc((size_t)Bsz * NI * Hh * 4);
    float* mWm_pp  = (float*)alloc((size_t)Bsz * NP * Hh * 4);
    float* qW_pool = (float*)alloc((size_t)Bsz * 512 * 4);
    float* qW      = (float*)alloc((size_t)Bsz * 2048 * 4);
    float* sc_pool = (float*)alloc((size_t)Bsz * NI * 4);
    float* scores  = (float*)alloc((size_t)Bsz * (NI + NP) * 4);
    float* gbuf    = (float*)alloc((size_t)Bsz * G4 * 4);
    float* c_vis   = (float*)alloc((size_t)Bsz * Hh * 4);
    float* c_lang  = (float*)alloc((size_t)Bsz * Hh * 4);
    bf16* x_vis    = (bf16*)alloc((size_t)Bsz * 2048 * 2);  // [img_pool|emb|h_vis]
    bf16* x_lang   = (bf16*)alloc((size_t)Bsz * 2560 * 2);  // [att|h_vis|h_lang]
    float* logits  = (float*)alloc((size_t)Bsz * Vv * 4);

    // ---- one-time: weight conversion / packing ----
    int n;
    n = G4 * 2048;
    hipLaunchKernelGGL(pack2_bf16_kernel, g1(n), dim3(256), 0, stream, vis_Wih, 1024, vis_Whh, 1024, wc_vis, G4);
    n = G4 * 2560;
    hipLaunchKernelGGL(pack2_bf16_kernel, g1(n), dim3(256), 0, stream, lang_Wih, 1536, lang_Whh, 1024, wc_lang, G4);
    n = Vv * Hh;
    hipLaunchKernelGGL(transpose_to_bf16_kernel, g1(n), dim3(256), 0, stream, proj_W, w_projT, Hh, Vv);
    n = Hh * Hh;
    hipLaunchKernelGGL(transpose_to_bf16_kernel, g1(n), dim3(256), 0, stream, img_Wq, w_qT, Hh, Hh);
    hipLaunchKernelGGL(transpose_to_bf16_kernel, g1(n), dim3(256), 0, stream, ppl_Wq, w_qT + (size_t)Hh * Hh, Hh, Hh);
    n = 512 * 512;
    hipLaunchKernelGGL(transpose_to_bf16_kernel, g1(n), dim3(256), 0, stream, pool_Wq, w_plWqT, 512, 512);
    hipLaunchKernelGGL(transpose_to_bf16_kernel, g1(n), dim3(256), 0, stream, pool_Wm, w_plWmT, 512, 512);
    n = 512 * Hh;
    hipLaunchKernelGGL(transpose_to_bf16_kernel, g1(n), dim3(256), 0, stream, img_Wm, w_imWmT, 512, Hh);
    hipLaunchKernelGGL(transpose_to_bf16_kernel, g1(n), dim3(256), 0, stream, ppl_Wm, w_ppWmT, 512, Hh);
    hipLaunchKernelGGL(concat2_f32_kernel, g1(2048), dim3(256), 0, stream, img_b, Hh, ppl_b, Hh, qb);

    // ---- one-time: activation conversion ----
    n = Bsz * NI * DIc;
    hipLaunchKernelGGL(f32_to_bf16_kernel, g1(n), dim3(256), 0, stream, img_feats, a_img, n);
    n = Bsz * NP * DIc;
    hipLaunchKernelGGL(f32_to_bf16_kernel, g1(n), dim3(256), 0, stream, ppl_feats, a_ppl, n);
    n = Bsz * 512;
    hipLaunchKernelGGL(f32_to_bf16_kernel, g1(n), dim3(256), 0, stream, answer, a_hint, n);

    // ---- zero state (c, and h slices inside x_vis/x_lang) ----
    hipLaunchKernelGGL(zero_u32_kernel, g1(Bsz * Hh), dim3(256), 0, stream, (unsigned*)c_vis, Bsz * Hh);
    hipLaunchKernelGGL(zero_u32_kernel, g1(Bsz * Hh), dim3(256), 0, stream, (unsigned*)c_lang, Bsz * Hh);
    hipLaunchKernelGGL(zero_u32_kernel, g1(Bsz * 1024), dim3(256), 0, stream, (unsigned*)x_vis, Bsz * 1024);
    hipLaunchKernelGGL(zero_u32_kernel, g1(Bsz * 1280), dim3(256), 0, stream, (unsigned*)x_lang, Bsz * 1280);

    // ---- one-time: hoisted attention keys (m @ Wm) ----
    launch_gemm(stream, a_hint, 512, w_plWqT, pool_b, qW_pool, 512, Bsz, 512, 512);
    launch_gemm(stream, a_img, 512, w_plWmT, nullptr, mWm_pl, 512, Bsz * NI, 512, 512);
    launch_gemm(stream, a_img, 512, w_imWmT, nullptr, mWm_im, Hh, Bsz * NI, Hh, 512);
    launch_gemm(stream, a_ppl, 512, w_ppWmT, nullptr, mWm_pp, Hh, Bsz * NP, Hh, 512);

    // ---- one-time: pooled image attention -> x_vis[:,0:512] ----
    hipLaunchKernelGGL(pool_scores_kernel, dim3(Bsz * NI), dim3(256), 0, stream, qW_pool, mWm_pl, pool_v, sc_pool);
    hipLaunchKernelGGL(softmax_kernel, dim3(Bsz), dim3(128), 0, stream, sc_pool, NI, NI);
    hipLaunchKernelGGL(ctx_pool_kernel, dim3(Bsz), dim3(256), 0, stream, sc_pool, img_feats, x_vis);

    // ---- recurrent steps ----
    for (int t = 0; t < Tt; ++t) {
        hipLaunchKernelGGL(embed_kernel, dim3(Bsz), dim3(256), 0, stream, question, emb, x_vis, t);
        // vis LSTM gates: [x_vis|h_vis] x [Wih|Whh]^T
        launch_gemm(stream, x_vis, 2048, wc_vis, vis_b, gbuf, G4, Bsz, G4, 2048);
        hipLaunchKernelGGL(lstm_elem_kernel, g1(Bsz * Hh), dim3(256), 0, stream,
                           gbuf, c_vis, x_vis + 1024, 2048, x_lang + 512, 2560);
        // attention queries: h_vis x [imgWq|pplWq]  (biases folded)
        launch_gemm(stream, x_vis + 1024, 2048, w_qT, qb, qW, 2048, Bsz, 2048, Hh);
        hipLaunchKernelGGL(scores_kernel, dim3(Bsz * (NI + NP)), dim3(256), 0, stream,
                           qW, mWm_im, mWm_pp, img_v, ppl_v, scores);
        hipLaunchKernelGGL(softmax_kernel, dim3(Bsz), dim3(128), 0, stream, scores, NI + NP, NI);
        hipLaunchKernelGGL(softmax_kernel, dim3(Bsz), dim3(128), 0, stream, scores + NI, NI + NP, NP);
        hipLaunchKernelGGL(ctx_main_kernel, dim3(Bsz), dim3(256), 0, stream,
                           scores, img_feats, ppl_feats, x_lang);
        // lang LSTM gates: [att|h_vis|h_lang] x [Wih|Whh]^T
        launch_gemm(stream, x_lang, 2560, wc_lang, lang_b, gbuf, G4, Bsz, G4, 2560);
        hipLaunchKernelGGL(lstm_elem_kernel, g1(Bsz * Hh), dim3(256), 0, stream,
                           gbuf, c_lang, x_lang + 1536, 2560, (bf16*)nullptr, 0);
        // vocab projection + log_softmax + argmax
        launch_gemm(stream, x_lang + 1536, 2560, w_projT, proj_b, logits, Vv, Bsz, Vv, Hh);
        hipLaunchKernelGGL(logsoftmax_kernel, dim3(Bsz), dim3(256), 0, stream, logits, out, t);
    }
}
